// MixtureOfExperts_29626684407896
// MI455X (gfx1250) — matrix operand order repaired and checked by reading, not compiled, not run on hardware
//
#include <hip/hip_runtime.h>
#include <stdint.h>

#define N_TOK 16384
#define DDIM  1024
#define NEXP  16
#define SCALE_F 32.0f          // sqrt(1024)
#define MT    64               // tokens per expert tile
#define AST   1032             // padded LDS row stride (bf16 elems), breaks bank conflicts

typedef __attribute__((ext_vector_type(16))) __bf16 v16bf;
typedef __attribute__((ext_vector_type(8)))  float  v8f;

union Frag16 {
  uint4 q[2];
  unsigned short u[16];
  v16bf v;
};

__device__ __forceinline__ unsigned short f2bf(float f) {
  unsigned int u = __float_as_uint(f);
  u = (u + 0x7FFFu + ((u >> 16) & 1u)) >> 16;   // RNE
  return (unsigned short)u;
}

// ---------------------------------------------------------------- zero helpers
__global__ void zero_f4_kernel(float4* __restrict__ p, long n4) {
  long i = (long)blockIdx.x * blockDim.x + threadIdx.x;
  for (; i < n4; i += (long)gridDim.x * blockDim.x) p[i] = make_float4(0.f, 0.f, 0.f, 0.f);
}
__global__ void zero_stats_kernel(int* counts, float* imp, float* loadv) {
  int t = threadIdx.x;
  if (t < NEXP) { counts[t] = 0; imp[t] = 0.f; loadv[t] = 0.f; }
}

// ---------------------------------------------------------------- gating
__global__ void gate_kernel(const float* __restrict__ x, const float* __restrict__ wg,
                            float* __restrict__ gates_out, float* __restrict__ inv_norm,
                            int* __restrict__ counts, int* __restrict__ lists,
                            float* __restrict__ gatev,
                            float* __restrict__ importance, float* __restrict__ loadv) {
  __shared__ float wgs[DDIM * NEXP];    // 64 KB
  const int tid = threadIdx.x;
  for (int i = tid; i < DDIM * NEXP; i += 256) wgs[i] = wg[i];
  __syncthreads();

  const int n = blockIdx.x * 256 + tid;
  float acc[NEXP];
#pragma unroll
  for (int e = 0; e < NEXP; ++e) acc[e] = 0.f;
  float ss = 0.f;
  const float4* xr = (const float4*)(x + (size_t)n * DDIM);
  for (int d4 = 0; d4 < DDIM / 4; ++d4) {
    float4 xv = xr[d4];
    float xs[4] = {xv.x, xv.y, xv.z, xv.w};
#pragma unroll
    for (int j = 0; j < 4; ++j) {
      float v = xs[j];
      ss += v * v;
      const float* wrow = &wgs[(d4 * 4 + j) * NEXP];
#pragma unroll
      for (int e = 0; e < NEXP; ++e) acc[e] += v * wrow[e];
    }
  }
  inv_norm[n] = 1.0f / fmaxf(sqrtf(ss), 1e-12f);

  float mx = acc[0];
#pragma unroll
  for (int e = 1; e < NEXP; ++e) mx = fmaxf(mx, acc[e]);
  float sum = 0.f;
#pragma unroll
  for (int e = 0; e < NEXP; ++e) { acc[e] = __expf(acc[e] - mx); sum += acc[e]; }
  float rs = 1.0f / sum;
#pragma unroll
  for (int e = 0; e < NEXP; ++e) acc[e] *= rs;

  int i1 = 0; float v1 = acc[0];
#pragma unroll
  for (int e = 1; e < NEXP; ++e) if (acc[e] > v1) { v1 = acc[e]; i1 = e; }
  int i2 = (i1 == 0) ? 1 : 0; float v2 = acc[i2];
#pragma unroll
  for (int e = 0; e < NEXP; ++e) if (e != i1 && acc[e] > v2) { v2 = acc[e]; i2 = e; }
  float gs = 1.0f / (v1 + v2 + 1e-6f);
  float g1 = v1 * gs, g2 = v2 * gs;

  float* grow = gates_out + (size_t)n * NEXP;
#pragma unroll
  for (int e = 0; e < NEXP; ++e) grow[e] = (e == i1) ? g1 : ((e == i2) ? g2 : 0.f);

  atomicAdd(&importance[i1], g1); atomicAdd(&loadv[i1], 1.0f);
  atomicAdd(&importance[i2], g2); atomicAdd(&loadv[i2], 1.0f);
  int p1 = atomicAdd(&counts[i1], 1);
  lists[i1 * N_TOK + p1] = n; gatev[i1 * N_TOK + p1] = g1;
  int p2 = atomicAdd(&counts[i2], 1);
  lists[i2 * N_TOK + p2] = n; gatev[i2 * N_TOK + p2] = g2;
}

// ---------------------------------------------------------------- x -> bf16 (normalized, * sqrt(D))
__global__ void xb_kernel(const float* __restrict__ x, const float* __restrict__ inv_norm,
                          unsigned short* __restrict__ xb) {
  size_t i = ((size_t)blockIdx.x * blockDim.x + threadIdx.x) * 4;
  if (i >= (size_t)N_TOK * DDIM) return;
  float s = inv_norm[i / DDIM] * SCALE_F;
  float4 v = *(const float4*)(x + i);
  unsigned short o[4] = {f2bf(v.x * s), f2bf(v.y * s), f2bf(v.z * s), f2bf(v.w * s)};
  *(uint2*)(xb + i) = *(uint2*)o;
}

// ---------------------------------------------------------------- weight bf16 prep
// W1b[e][n][k] = bf16(W1[e][n][k] * gamma[e][k] * SCALE)   (gamma folded in)
__global__ void w1b_kernel(const float* __restrict__ W1, const float* __restrict__ gamma,
                           unsigned short* __restrict__ W1b) {
  size_t i = ((size_t)blockIdx.x * blockDim.x + threadIdx.x) * 8;
  if (i >= (size_t)NEXP * 2 * DDIM * DDIM) return;
  int e = (int)(i / ((size_t)2 * DDIM * DDIM));
  int k = (int)(i % DDIM);
  const float* g = gamma + (size_t)e * DDIM + k;
  float4 a = *(const float4*)(W1 + i);
  float4 b = *(const float4*)(W1 + i + 4);
  float4 ga = *(const float4*)g;
  float4 gb = *(const float4*)(g + 4);
  unsigned short o[8] = {
    f2bf(a.x * ga.x * SCALE_F), f2bf(a.y * ga.y * SCALE_F),
    f2bf(a.z * ga.z * SCALE_F), f2bf(a.w * ga.w * SCALE_F),
    f2bf(b.x * gb.x * SCALE_F), f2bf(b.y * gb.y * SCALE_F),
    f2bf(b.z * gb.z * SCALE_F), f2bf(b.w * gb.w * SCALE_F)};
  *(uint4*)(W1b + i) = *(uint4*)o;
}
__global__ void w2b_kernel(const float* __restrict__ W2, unsigned short* __restrict__ W2b) {
  size_t i = ((size_t)blockIdx.x * blockDim.x + threadIdx.x) * 8;
  if (i >= (size_t)NEXP * DDIM * DDIM) return;
  float4 a = *(const float4*)(W2 + i);
  float4 b = *(const float4*)(W2 + i + 4);
  unsigned short o[8] = {f2bf(a.x), f2bf(a.y), f2bf(a.z), f2bf(a.w),
                         f2bf(b.x), f2bf(b.y), f2bf(b.z), f2bf(b.w)};
  *(uint4*)(W2b + i) = *(uint4*)o;
}

// ---------------------------------------------------------------- tile scheduler (device-side, graph-safe)
__global__ void tiles_kernel(const int* __restrict__ counts, int* __restrict__ tileExpert,
                             int* __restrict__ tileMbase, int* __restrict__ totalTiles) {
  if (threadIdx.x == 0 && blockIdx.x == 0) {
    int tt = 0;
    for (int e = 0; e < NEXP; ++e) {
      int c = counts[e];
      int nt = (c + MT - 1) / MT;
      for (int j = 0; j < nt; ++j) { tileExpert[tt] = e; tileMbase[tt] = j * MT; ++tt; }
    }
    *totalTiles = tt;
  }
}

// ---------------------------------------------------------------- fused expert kernel
__global__ void __launch_bounds__(256)
expert_kernel(const unsigned short* __restrict__ xb,
              const unsigned short* __restrict__ W1b, const float* __restrict__ b1,
              const float* __restrict__ mult,
              const unsigned short* __restrict__ W2b, const float* __restrict__ b2,
              const int* __restrict__ counts, const int* __restrict__ lists,
              const float* __restrict__ gatev,
              const int* __restrict__ tileExpert, const int* __restrict__ tileMbase,
              const int* __restrict__ totalTiles, float* __restrict__ y) {
  extern __shared__ unsigned char smem_raw[];
  unsigned short* As = (unsigned short*)smem_raw;          // 64 x 1032 bf16 = 132096 B
  unsigned short* Us = As + MT * AST;                      // 64 x 1032 bf16 = 132096 B
  int*   tokS  = (int*)(Us + MT * AST);                    // 64 ints
  float* gateS = (float*)(tokS + MT);                      // 64 floats

  const int tid  = threadIdx.x;
  const int lane = tid & 31, wave = tid >> 5;              // 8 waves (wave32)
  const int ln   = lane & 15, hf = lane >> 4;
  const int TT   = *totalTiles;

  for (int t = blockIdx.x; t < TT; t += gridDim.x) {
    const int e  = tileExpert[t];
    const int m0 = tileMbase[t];
    const int ce = counts[e];

    if (tid < MT) {
      int idx = m0 + tid;
      if (idx < ce) { tokS[tid] = lists[e * N_TOK + idx]; gateS[tid] = gatev[e * N_TOK + idx]; }
      else          { tokS[tid] = -1;                     gateS[tid] = 0.f; }
    }
    __syncthreads();

    // ---- stage A tile via CDNA5 async gather copy: LDS[As + m*AST + kb] = xb[tok[m]][kb..kb+7]
    {
      const int t0 = tokS[0];  // first token of tile always valid; padded rows read it (outputs discarded)
      for (int c = tid; c < MT * (DDIM / 8); c += 256) {
        int m  = c >> 7;             // DDIM/8 == 128 16B-granules per row
        int kb = (c & 127) * 8;
        int tok = tokS[m]; if (tok < 0) tok = t0;
        unsigned ldsa = (unsigned)(uintptr_t)(As + m * AST + kb);
        unsigned long long ga = (unsigned long long)(uintptr_t)(xb + (size_t)tok * DDIM + kb);
        asm volatile("global_load_async_to_lds_b128 %0, %1, off"
                     :: "v"(ldsa), "v"(ga) : "memory");
      }
      asm volatile("s_wait_asynccnt 0x0" ::: "memory");
    }
    __syncthreads();

    const unsigned short* W1e = W1b + (size_t)e * 2 * DDIM * DDIM;
    const float* b1e = b1 + (size_t)e * 2 * DDIM;
    const float* me  = mult + (size_t)e * DDIM;

    // ---- GEMM1 + gated GELU -> Us. Each wave owns paired (value, gate) 16-col chunks.
    for (int cc = wave; cc < DDIM / 16; cc += 8) {
      const int colV = cc * 16;
      const int colG = DDIM + cc * 16;
      v8f cV[4], cG[4];
      const float bv = b1e[colV + ln];
      const float bg = b1e[colG + ln];
#pragma unroll
      for (int ms = 0; ms < 4; ++ms)
#pragma unroll
        for (int v = 0; v < 8; ++v) { cV[ms][v] = bv; cG[ms][v] = bg; }

      const unsigned short* bpV = W1e + (size_t)(colV + ln) * DDIM + hf * 16;
      const unsigned short* bpG = W1e + (size_t)(colG + ln) * DDIM + hf * 16;

      for (int kb = 0; kb < DDIM; kb += 32) {
        Frag16 bV, bG;
        bV.q[0] = *(const uint4*)(bpV + kb);
        bV.q[1] = *(const uint4*)(bpV + kb + 8);
        bG.q[0] = *(const uint4*)(bpG + kb);
        bG.q[1] = *(const uint4*)(bpG + kb + 8);
#pragma unroll
        for (int ms = 0; ms < 4; ++ms) {
          Frag16 a;
          const unsigned short* rp = As + (ms * 16 + ln) * AST + kb + hf * 16;
          a.q[0] = *(const uint4*)rp;            // k = kb + hf*16 .. +7
          a.q[1] = *(const uint4*)(rp + 8);      // k = kb + hf*16 + 8 .. +15
          cV[ms] = __builtin_amdgcn_wmma_f32_16x16x32_bf16(false, a.v, false, bV.v,
                                                           (short)0, cV[ms], false, false);
          cG[ms] = __builtin_amdgcn_wmma_f32_16x16x32_bf16(false, a.v, false, bG.v,
                                                           (short)0, cG[ms], false, false);
        }
      }
      const int col = colV + ln;
      const float mcol = me[col];
#pragma unroll
      for (int ms = 0; ms < 4; ++ms)
#pragma unroll
        for (int v = 0; v < 8; ++v) {
          int m = ms * 16 + hf * 8 + v;
          float hg = cG[ms][v];
          float gl = 0.5f * hg * (1.0f + erff(hg * 0.70710678118f));
          Us[m * AST + col] = f2bf(gl * cV[ms][v] * mcol);
        }
    }
    __syncthreads();

    // ---- GEMM2: y += gate * (U @ W2^T + b2)
    const unsigned short* W2e = W2b + (size_t)e * DDIM * DDIM;
    const float* b2e = b2 + (size_t)e * DDIM;
    for (int cc = wave; cc < DDIM / 16; cc += 8) {
      const int col0 = cc * 16;
      v8f c2[4];
      const float bb = b2e[col0 + ln];
#pragma unroll
      for (int ms = 0; ms < 4; ++ms)
#pragma unroll
        for (int v = 0; v < 8; ++v) c2[ms][v] = bb;

      const unsigned short* bp = W2e + (size_t)(col0 + ln) * DDIM + hf * 16;
      for (int kb = 0; kb < DDIM; kb += 32) {
        Frag16 b;
        b.q[0] = *(const uint4*)(bp + kb);
        b.q[1] = *(const uint4*)(bp + kb + 8);
#pragma unroll
        for (int ms = 0; ms < 4; ++ms) {
          Frag16 a;
          const unsigned short* rp = Us + (ms * 16 + ln) * AST + kb + hf * 16;
          a.q[0] = *(const uint4*)rp;
          a.q[1] = *(const uint4*)(rp + 8);
          c2[ms] = __builtin_amdgcn_wmma_f32_16x16x32_bf16(false, a.v, false, b.v,
                                                           (short)0, c2[ms], false, false);
        }
      }
      const int col = col0 + ln;
#pragma unroll
      for (int ms = 0; ms < 4; ++ms)
#pragma unroll
        for (int v = 0; v < 8; ++v) {
          int m = ms * 16 + hf * 8 + v;
          int tok = tokS[m];
          if (tok >= 0) atomicAdd(&y[(size_t)tok * DDIM + col], c2[ms][v] * gateS[m]);
        }
    }
    __syncthreads();
  }
}

// ---------------------------------------------------------------- aux loss
__global__ void loss_kernel(const float* __restrict__ imp, const float* __restrict__ loadv,
                            float* __restrict__ out_loss) {
  if (threadIdx.x == 0 && blockIdx.x == 0) {
    float cv[2];
    for (int s = 0; s < 2; ++s) {
      const float* v = s ? loadv : imp;
      float mean = 0.f;
      for (int e = 0; e < NEXP; ++e) mean += v[e];
      mean *= (1.0f / NEXP);
      float var = 0.f;
      for (int e = 0; e < NEXP; ++e) { float d = v[e] - mean; var += d * d; }
      var *= (1.0f / (NEXP - 1));
      cv[s] = var / (mean * mean + 1e-10f);
    }
    *out_loss = (cv[0] + cv[1]) * 0.01f;
  }
}

// ---------------------------------------------------------------- launcher
extern "C" void kernel_launch(void* const* d_in, const int* in_sizes, int n_in,
                              void* d_out, int out_size, void* d_ws, size_t ws_size,
                              hipStream_t stream) {
  (void)in_sizes; (void)n_in; (void)out_size; (void)ws_size;
  const float* x     = (const float*)d_in[0];
  const float* wgate = (const float*)d_in[1];
  const float* gamma = (const float*)d_in[2];
  const float* W1    = (const float*)d_in[3];
  const float* b1    = (const float*)d_in[4];
  const float* mult  = (const float*)d_in[5];
  const float* W2    = (const float*)d_in[6];
  const float* b2    = (const float*)d_in[7];

  float* y         = (float*)d_out;                       // N x D
  float* gates_out = y + (size_t)N_TOK * DDIM;            // N x E
  float* loss_out  = gates_out + (size_t)N_TOK * NEXP;    // scalar

  // workspace layout
  uint8_t* w = (uint8_t*)d_ws;
  size_t o = 0;
  auto al = [](size_t v) { return (v + 255) & ~(size_t)255; };
  unsigned short* xb  = (unsigned short*)(w + o); o = al(o + (size_t)N_TOK * DDIM * 2);
  unsigned short* W1b = (unsigned short*)(w + o); o = al(o + (size_t)NEXP * 2 * DDIM * DDIM * 2);
  unsigned short* W2b = (unsigned short*)(w + o); o = al(o + (size_t)NEXP * DDIM * DDIM * 2);
  float* inv_norm     = (float*)(w + o);          o = al(o + (size_t)N_TOK * 4);
  int*   counts       = (int*)(w + o);            o = al(o + NEXP * 4);
  float* imp          = (float*)(w + o);          o = al(o + NEXP * 4);
  float* loadv        = (float*)(w + o);          o = al(o + NEXP * 4);
  int*   totalTiles   = (int*)(w + o);            o = al(o + 4);
  int*   tileExpert   = (int*)(w + o);            o = al(o + 1024 * 4);
  int*   tileMbase    = (int*)(w + o);            o = al(o + 1024 * 4);
  int*   lists        = (int*)(w + o);            o = al(o + (size_t)NEXP * N_TOK * 4);
  float* gatev        = (float*)(w + o);          o = al(o + (size_t)NEXP * N_TOK * 4);

  zero_f4_kernel<<<4096, 256, 0, stream>>>((float4*)y, (long)N_TOK * DDIM / 4);
  zero_stats_kernel<<<1, 32, 0, stream>>>(counts, imp, loadv);

  // weight bf16 prep (gamma*sqrt(D) folded into W1b)
  w1b_kernel<<<16384, 256, 0, stream>>>(W1, gamma, W1b);
  w2b_kernel<<<8192, 256, 0, stream>>>(W2, W2b);

  gate_kernel<<<N_TOK / 256, 256, 0, stream>>>(x, wgate, gates_out, inv_norm,
                                               counts, lists, gatev, imp, loadv);
  xb_kernel<<<(N_TOK * DDIM / 4) / 256, 256, 0, stream>>>(x, inv_norm, xb);
  tiles_kernel<<<1, 32, 0, stream>>>(counts, tileExpert, tileMbase, totalTiles);

  const int smem = MT * AST * 2 * 2 + MT * 8;   // As + Us + tok/gate = 264704 B (<320KB WGP LDS)
  expert_kernel<<<544, 256, smem, stream>>>(xb, W1b, b1, mult, W2b, b2,
                                            counts, lists, gatev,
                                            tileExpert, tileMbase, totalTiles, y);

  loss_kernel<<<1, 32, 0, stream>>>(imp, loadv, loss_out);
}
